// SpindleResidual_9861244912120
// MI455X (gfx1250) — matrix operand order repaired and checked by reading, not compiled, not run on hardware
//
#include <hip/hip_runtime.h>
#include <hip/hip_bf16.h>

// SpindleResidual for MI455X (gfx1250, wave32, WMMA).
// Pipeline (all on `stream`, deterministic, no float atomics):
//  1) chan_sum(x)            -> s1[32][128]
//  2) rank_select 128->64    -> sel1
//  3) head_gemm (WMMA f16)   -> h_pre[32][256][4096]  (fp32 accum)
//  4) bn_stats(h_pre)        -> mean1/rstd1
//  5) dw 3x3 + BN1 + hswish fused on input side -> h2_pre
//  6) bn_stats(h2_pre)       -> mean2/rstd2
//  7) chan_sum(act2(h2_pre)) -> s2[32][256]
//  8) rank_select 256->128   -> sel2
//  9) tail_gemm (WMMA f16, act2 fused into B-frag load) -> t_pre (reuses h_pre slot)
// 10) bn_stats(t_pre)        -> mean3/rstd3
// 11) out = BN3(t_pre) + x
// Workspace: 2 * 128MiB big buffers + ~80KB small. (~256 MiB)

typedef _Float16 v16h __attribute__((ext_vector_type(16)));
typedef float    v8f  __attribute__((ext_vector_type(8)));

#define HW 4096
#define WDIM 64
#define BATCH 32
#define CIN 128
#define CMID 256
#define COUT 128
#define AHEAD 64
#define ATAIL 128

__device__ __forceinline__ float hswish(float v) {
    float t = fminf(fmaxf(v + 3.0f, 0.0f), 6.0f);
    return v * t * (1.0f / 6.0f);
}

// ---------------------------------------------------------------------------
// Per-(b,c) spatial sum, optionally through act(v) = hswish(v*sc + sh)
// grid.x = B*C, block = 256
__global__ __launch_bounds__(256) void chan_sum_kernel(
    const float* __restrict__ src, int C, int apply_act,
    const float* __restrict__ mean, const float* __restrict__ rstd,
    const float* __restrict__ g, const float* __restrict__ be,
    float* __restrict__ out)
{
    const int bc = blockIdx.x;
    const int c  = bc % C;
    const int tid = threadIdx.x;
    float scv = 1.0f, shv = 0.0f;
    if (apply_act) { scv = rstd[c] * g[c]; shv = be[c] - mean[c] * scv; }
    const float* p = src + (size_t)bc * HW;
    float s = 0.0f;
    for (int i = tid; i < HW; i += 256) {
        float v = p[i];
        if (apply_act) v = hswish(v * scv + shv);
        s += v;
    }
    __shared__ float ss[256];
    ss[tid] = s; __syncthreads();
    for (int o = 128; o > 0; o >>= 1) {
        if (tid < o) ss[tid] += ss[tid + o];
        __syncthreads();
    }
    if (tid == 0) out[bc] = ss[0];
}

// ---------------------------------------------------------------------------
// Stable ascending argsort (rank by count), keep first `keep`.
// grid.x = B, blockDim = C (<=256)
__global__ void rank_select_kernel(const float* __restrict__ s, int C, int keep,
                                   int* __restrict__ sel)
{
    __shared__ float sv[256];
    const int b = blockIdx.x;
    const int c = threadIdx.x;
    sv[c] = s[b * C + c];
    __syncthreads();
    const float my = sv[c];
    int rank = 0;
    for (int j = 0; j < C; ++j) {
        float o = sv[j];
        rank += (o < my) || (o == my && j < c) ? 1 : 0;
    }
    if (rank < keep) sel[b * keep + rank] = c;
}

// ---------------------------------------------------------------------------
// Head GEMM: h_pre[b][m][n] = sum_k w[m][k] * x[b][sel1[b][k]][n] + bias[m]
// M=256, K=64 (2 k-steps of 32), N=4096 per sample.
// grid = (64 n-chunks of 64, 2 m-blocks of 128, 32 batch), block = 256 (8 waves)
__global__ __launch_bounds__(256) void head_gemm_kernel(
    const float* __restrict__ x, const float* __restrict__ w,
    const float* __restrict__ bias, const int* __restrict__ sel,
    float* __restrict__ hpre)
{
    __shared__ int rows[AHEAD];
    const int b   = blockIdx.z;
    const int tid = threadIdx.x;
    if (tid < AHEAD) rows[tid] = sel[b * AHEAD + tid] * HW;
    __syncthreads();

    const int lane = tid & 31;
    const int wid  = tid >> 5;
    const int m0   = blockIdx.y * 128 + wid * 16;
    const int nbase = blockIdx.x * 64;
    const int hs = lane >> 4;        // half-wave select
    const int lm = lane & 15;

    // A fragments (weights, reused over 4 N tiles). contiguous k order used here (B matches):
    // lanes 0-15 hold K 0-15 ; lanes 16-31 hold K 16-31.
    v16h a0, a1;
    {
        const float* wr = w + (size_t)(m0 + lm) * AHEAD;
        const int kh = hs * 16;
        #pragma unroll
        for (int j = 0; j < 8; ++j) {
            a0[j]     = (_Float16)wr[kh + j];
            a0[8 + j] = (_Float16)wr[8 + kh + j];
            a1[j]     = (_Float16)wr[32 + kh + j];
            a1[8 + j] = (_Float16)wr[40 + kh + j];
        }
    }

    const float* xb = x + (size_t)b * CIN * HW;
    const int kb = hs * 16;          // B frag: lanes 0-15 K 0-15, lanes 16-31 K 16-31
    float* outb = hpre + (size_t)b * CMID * HW;

    for (int t = 0; t < 4; ++t) {
        const int n = nbase + t * 16 + lm;
        if (t < 3) __builtin_prefetch(xb + rows[kb] + nbase + (t + 1) * 16 + lm, 0, 1);
        v16h bf0, bf1;
        #pragma unroll
        for (int j = 0; j < 16; ++j) {
            bf0[j] = (_Float16)xb[rows[kb + j] + n];
            bf1[j] = (_Float16)xb[rows[32 + kb + j] + n];
        }
        v8f c = {};
        c = __builtin_amdgcn_wmma_f32_16x16x32_f16(false, a0, false, bf0,
                                                   (short)0, c, false, false);
        c = __builtin_amdgcn_wmma_f32_16x16x32_f16(false, a1, false, bf1,
                                                   (short)0, c, false, false);
        #pragma unroll
        for (int r = 0; r < 8; ++r) {
            const int m = m0 + r + 8 * hs;   // D layout: VGPR r -> M = r (+8 hi half)
            outb[(size_t)m * HW + n] = c[r] + bias[m];
        }
    }
}

// ---------------------------------------------------------------------------
// Training-mode BN stats per channel over (N,H,W): biased var, rstd.
// grid.x = C, block = 256
__global__ __launch_bounds__(256) void bn_stats_kernel(
    const float* __restrict__ src, int C,
    float* __restrict__ mean, float* __restrict__ rstd)
{
    const int c = blockIdx.x;
    const int tid = threadIdx.x;
    float s = 0.0f, q = 0.0f;
    for (int b = 0; b < BATCH; ++b) {
        const float* p = src + ((size_t)b * C + c) * HW;
        for (int i = tid; i < HW; i += 256) {
            float v = p[i];
            s += v; q += v * v;
        }
    }
    __shared__ float ss[256], sq[256];
    ss[tid] = s; sq[tid] = q; __syncthreads();
    for (int o = 128; o > 0; o >>= 1) {
        if (tid < o) { ss[tid] += ss[tid + o]; sq[tid] += sq[tid + o]; }
        __syncthreads();
    }
    if (tid == 0) {
        const float inv = 1.0f / (float)(BATCH * HW);
        float m = ss[0] * inv;
        float v = sq[0] * inv - m * m;
        mean[c] = m;
        rstd[c] = rsqrtf(v + 1e-5f);
    }
}

// ---------------------------------------------------------------------------
// Depthwise 3x3 (cross-correlation, SAME zero-pad of the *activated* input)
// with BN1+hardswish fused on the input side, + per-channel bias.
// grid = (16 pixel chunks, 256 channels, 32 batch), block = 256
__global__ __launch_bounds__(256) void dw_body_kernel(
    const float* __restrict__ hpre, const float* __restrict__ wb,
    const float* __restrict__ bb,
    const float* __restrict__ mean1, const float* __restrict__ rstd1,
    const float* __restrict__ g1, const float* __restrict__ be1,
    float* __restrict__ h2)
{
    const int c = blockIdx.y;
    const int b = blockIdx.z;
    const int p = blockIdx.x * 256 + threadIdx.x;
    const int y = p >> 6;
    const int xx = p & (WDIM - 1);
    const float scv = rstd1[c] * g1[c];
    const float shv = be1[c] - mean1[c] * scv;
    const float* src = hpre + ((size_t)b * CMID + c) * HW;
    const float* wc = wb + c * 9;
    float acc = bb[c];
    #pragma unroll
    for (int dy = -1; dy <= 1; ++dy) {
        #pragma unroll
        for (int dx = -1; dx <= 1; ++dx) {
            int yy = y + dy, xc = xx + dx;
            if (yy >= 0 && yy < WDIM && xc >= 0 && xc < WDIM) {
                float v = src[yy * WDIM + xc];
                v = hswish(v * scv + shv);
                acc += v * wc[(dy + 1) * 3 + (dx + 1)];
            }
        }
    }
    h2[((size_t)b * CMID + c) * HW + p] = acc;
}

// ---------------------------------------------------------------------------
// Tail GEMM: t_pre[b][m][n] = sum_k w[m][k] * act2(h2[b][sel2[b][k]][n]) + bias[m]
// M=128, K=128 (4 k-steps of 32), N=4096. act2 fused into B-fragment loads.
// grid = (64 n-chunks, 1, 32 batch), block = 256 (8 waves cover all 128 M rows)
__global__ __launch_bounds__(256) void tail_gemm_kernel(
    const float* __restrict__ h2, const float* __restrict__ w,
    const float* __restrict__ bias, const int* __restrict__ sel,
    const float* __restrict__ mean2, const float* __restrict__ rstd2,
    const float* __restrict__ g2, const float* __restrict__ be2,
    float* __restrict__ tpre)
{
    __shared__ int   rows[ATAIL];
    __shared__ float sc[ATAIL], sh[ATAIL];
    const int b   = blockIdx.z;
    const int tid = threadIdx.x;
    if (tid < ATAIL) {
        const int ch = sel[b * ATAIL + tid];
        rows[tid] = ch * HW;
        const float s = rstd2[ch] * g2[ch];
        sc[tid] = s;
        sh[tid] = be2[ch] - mean2[ch] * s;
    }
    __syncthreads();

    const int lane = tid & 31;
    const int wid  = tid >> 5;
    const int m0   = wid * 16;
    const int nbase = blockIdx.x * 64;
    const int hs = lane >> 4;
    const int lm = lane & 15;

    v16h a[4];
    {
        const float* wr = w + (size_t)(m0 + lm) * ATAIL;
        const int kh = hs * 16;
        #pragma unroll
        for (int ks = 0; ks < 4; ++ks) {
            const int base = ks * 32;
            #pragma unroll
            for (int j = 0; j < 8; ++j) {
                a[ks][j]     = (_Float16)wr[base + kh + j];
                a[ks][8 + j] = (_Float16)wr[base + 8 + kh + j];
            }
        }
    }

    const float* hb = h2 + (size_t)b * CMID * HW;
    const int kb = hs * 16;
    float* outb = tpre + (size_t)b * COUT * HW;

    for (int t = 0; t < 4; ++t) {
        const int n = nbase + t * 16 + lm;
        if (t < 3) __builtin_prefetch(hb + rows[kb] + nbase + (t + 1) * 16 + lm, 0, 1);
        v8f c = {};
        #pragma unroll
        for (int ks = 0; ks < 4; ++ks) {
            v16h bf;
            #pragma unroll
            for (int j = 0; j < 16; ++j) {
                const int k = ks * 32 + kb + j;
                float v = hb[rows[k] + n];
                v = hswish(v * sc[k] + sh[k]);
                bf[j] = (_Float16)v;
            }
            c = __builtin_amdgcn_wmma_f32_16x16x32_f16(false, a[ks], false, bf,
                                                       (short)0, c, false, false);
        }
        #pragma unroll
        for (int r = 0; r < 8; ++r) {
            const int m = m0 + r + 8 * hs;
            outb[(size_t)m * HW + n] = c[r] + bias[m];
        }
    }
}

// ---------------------------------------------------------------------------
// out = BN3(t_pre) + x     (grid = 32*128*4096/256)
__global__ __launch_bounds__(256) void final_kernel(
    const float* __restrict__ tpre, const float* __restrict__ x,
    const float* __restrict__ mean3, const float* __restrict__ rstd3,
    const float* __restrict__ g3, const float* __restrict__ be3,
    float* __restrict__ out)
{
    const size_t i = (size_t)blockIdx.x * 256 + threadIdx.x;
    const int c = (int)((i >> 12) & (COUT - 1));
    const float s = rstd3[c] * g3[c];
    out[i] = (tpre[i] - mean3[c]) * s + be3[c] + x[i];
}

// ---------------------------------------------------------------------------
extern "C" void kernel_launch(void* const* d_in, const int* in_sizes, int n_in,
                              void* d_out, int out_size, void* d_ws, size_t ws_size,
                              hipStream_t stream)
{
    const float* x      = (const float*)d_in[0];
    const float* w_head = (const float*)d_in[1];
    const float* b_head = (const float*)d_in[2];
    const float* g1     = (const float*)d_in[3];
    const float* be1    = (const float*)d_in[4];
    const float* w_body = (const float*)d_in[5];
    const float* b_body = (const float*)d_in[6];
    const float* g2     = (const float*)d_in[7];
    const float* be2    = (const float*)d_in[8];
    const float* w_tail = (const float*)d_in[9];
    const float* b_tail = (const float*)d_in[10];
    const float* g3     = (const float*)d_in[11];
    const float* be3    = (const float*)d_in[12];
    float* out = (float*)d_out;

    // workspace layout (floats)
    const size_t BIG = (size_t)BATCH * CMID * HW;   // 33554432 floats = 128 MiB
    float* ws    = (float*)d_ws;
    float* hpre  = ws;                // [BIG]
    float* h2    = ws + BIG;          // [BIG]
    float* tpre  = ws;                // reuses hpre slot (hpre dead after dw pass)
    float* sm    = ws + 2 * BIG;
    float* s1    = sm;                // 4096
    float* s2    = sm + 4096;         // 8192
    float* mean1 = sm + 12288;  float* rstd1 = mean1 + 256;
    float* mean2 = rstd1 + 256; float* rstd2 = mean2 + 256;
    float* mean3 = rstd2 + 256; float* rstd3 = mean3 + 128;
    int*   sel1  = (int*)(rstd3 + 128);   // 32*64
    int*   sel2  = sel1 + BATCH * AHEAD;  // 32*128

    // 1) head PCA sums on x
    chan_sum_kernel<<<BATCH * CIN, 256, 0, stream>>>(
        x, CIN, 0, nullptr, nullptr, nullptr, nullptr, s1);
    // 2) stable rank -> keep first 64 channels per sample
    rank_select_kernel<<<BATCH, CIN, 0, stream>>>(s1, CIN, AHEAD, sel1);
    // 3) head GEMM (WMMA)
    head_gemm_kernel<<<dim3(64, 2, BATCH), 256, 0, stream>>>(
        x, w_head, b_head, sel1, hpre);
    // 4) BN1 stats
    bn_stats_kernel<<<CMID, 256, 0, stream>>>(hpre, CMID, mean1, rstd1);
    // 5) depthwise 3x3 with fused BN1+hardswish input
    dw_body_kernel<<<dim3(16, CMID, BATCH), 256, 0, stream>>>(
        hpre, w_body, b_body, mean1, rstd1, g1, be1, h2);
    // 6) BN2 stats
    bn_stats_kernel<<<CMID, 256, 0, stream>>>(h2, CMID, mean2, rstd2);
    // 7) tail PCA sums on act2(h2)
    chan_sum_kernel<<<BATCH * CMID, 256, 0, stream>>>(
        h2, CMID, 1, mean2, rstd2, g2, be2, s2);
    // 8) stable rank -> keep first 128 channels per sample
    rank_select_kernel<<<BATCH, CMID, 0, stream>>>(s2, CMID, ATAIL, sel2);
    // 9) tail GEMM (WMMA, act2 fused)
    tail_gemm_kernel<<<dim3(64, 1, BATCH), 256, 0, stream>>>(
        h2, w_tail, b_tail, sel2, mean2, rstd2, g2, be2, tpre);
    // 10) BN3 stats
    bn_stats_kernel<<<COUT, 256, 0, stream>>>(tpre, COUT, mean3, rstd3);
    // 11) BN3 affine + residual
    final_kernel<<<(BATCH * COUT * HW) / 256, 256, 0, stream>>>(
        tpre, x, mean3, rstd3, g3, be3, out);
}
